// CrossCBR_26010321944725
// MI455X (gfx1250) — matrix operand order repaired and checked by reading, not compiled, not run on hardware
//
#include <hip/hip_runtime.h>
#include <hip/hip_bf16.h>
#include <math.h>

// ---------------- problem constants (match reference) ----------------
#define NU   50000
#define NBU  20000
#define NI   100000
#define DD   64
#define NIL  (NU + NI)    // 150000 nodes in user-item graph
#define NBL  (NU + NBU)   // 70000 nodes in user-bundle graph
#define BB   2048         // batch
#define INV_CTEMP 4.0f    // 1 / 0.25

typedef __attribute__((ext_vector_type(16))) _Float16 v16h;
typedef __attribute__((ext_vector_type(8)))  float    v8f;

// Hardware f32 atomic add (global_atomic_add_f32, non-returning, agent scope).
__device__ __forceinline__ void atomic_add_f32(float* p, float v) {
  __hip_atomic_fetch_add(p, v, __ATOMIC_RELAXED, __HIP_MEMORY_SCOPE_AGENT);
}

// ---------------- degree / normalization helpers ----------------
__global__ void k_deg_pair(const int* __restrict__ a, const int* __restrict__ b,
                           int E, int off, float* __restrict__ deg) {
  int e = blockIdx.x * blockDim.x + threadIdx.x;
  if (e >= E) return;
  atomic_add_f32(&deg[a[e]], 1.0f);
  atomic_add_f32(&deg[b[e] + off], 1.0f);
}

__global__ void k_deg_single(const int* __restrict__ a, int E, float* __restrict__ deg) {
  int e = blockIdx.x * blockDim.x + threadIdx.x;
  if (e >= E) return;
  atomic_add_f32(&deg[a[e]], 1.0f);
}

__global__ void k_dinv(float* __restrict__ d, int n) {
  int i = blockIdx.x * blockDim.x + threadIdx.x;
  if (i >= n) return;
  d[i] = 1.0f / (sqrtf(d[i]) + 1e-8f);
}

__global__ void k_winv(float* __restrict__ d, int n) {
  int i = blockIdx.x * blockDim.x + threadIdx.x;
  if (i >= n) return;
  d[i] = 1.0f / (d[i] + 1e-8f);
}

// ---------------- SpMM: one wave32 per undirected edge, float2 per lane ----------------
__global__ void k_spmm(const int* __restrict__ eu, const int* __restrict__ ev,
                       int E, int off, const float* __restrict__ dinv,
                       const float* __restrict__ fin, float* __restrict__ g) {
  int gid  = blockIdx.x * blockDim.x + threadIdx.x;
  int e    = gid >> 5;
  int lane = gid & 31;
  if (e >= E) return;
  int u = eu[e];
  int v = ev[e] + off;
  float val = dinv[u] * dinv[v];
  const float2* f2 = (const float2*)fin;
  float2 fu = f2[(size_t)u * 32 + lane];
  float2 fv = f2[(size_t)v * 32 + lane];
  atomic_add_f32(&g[(size_t)u * DD + 2 * lane    ], val * fv.x);
  atomic_add_f32(&g[(size_t)u * DD + 2 * lane + 1], val * fv.y);
  atomic_add_f32(&g[(size_t)v * DD + 2 * lane    ], val * fu.x);
  atomic_add_f32(&g[(size_t)v * DD + 2 * lane + 1], val * fu.y);
}

// f = g * s; acc += f / max(||f||, 1e-12).  One wave per row.
__global__ void k_row_update(const float* __restrict__ g, float* __restrict__ f,
                             float* __restrict__ acc, int n, float s) {
  int row  = (blockIdx.x * blockDim.x + threadIdx.x) >> 5;
  int lane = threadIdx.x & 31;
  if (row >= n) return;
  float2 v = ((const float2*)g)[(size_t)row * 32 + lane];
  v.x *= s; v.y *= s;
  float ss = v.x * v.x + v.y * v.y;
  #pragma unroll
  for (int m = 16; m > 0; m >>= 1) ss += __shfl_xor(ss, m, 32);
  float inv = 1.0f / fmaxf(sqrtf(ss), 1e-12f);
  ((float2*)f)[(size_t)row * 32 + lane] = v;
  float2 a = ((float2*)acc)[(size_t)row * 32 + lane];
  a.x += v.x * inv; a.y += v.y * inv;
  ((float2*)acc)[(size_t)row * 32 + lane] = a;
}

// ---------------- bundle mean-aggregation of item embeddings ----------------
__global__ void k_bundle_agg(const int* __restrict__ bb, const int* __restrict__ bi,
                             int E, const float* __restrict__ winv,
                             const float* __restrict__ items, float* __restrict__ outb) {
  int gid  = blockIdx.x * blockDim.x + threadIdx.x;
  int e    = gid >> 5;
  int lane = gid & 31;
  if (e >= E) return;
  int b  = bb[e];
  int it = bi[e];
  float w = winv[b];
  float2 x = ((const float2*)items)[(size_t)it * 32 + lane];
  atomic_add_f32(&outb[(size_t)b * DD + 2 * lane    ], w * x.x);
  atomic_add_f32(&outb[(size_t)b * DD + 2 * lane + 1], w * x.y);
}

// ---------------- BPR loss ----------------
__global__ void k_bpr(const int* __restrict__ users, const int* __restrict__ bundles,
                      const float* __restrict__ acc_il, const float* __restrict__ ilb,
                      const float* __restrict__ acc_bl, int B, float* __restrict__ s_bpr) {
  int r = blockIdx.x * blockDim.x + threadIdx.x;
  if (r >= B) return;
  int u  = users[r];
  int b0 = bundles[2 * r], b1 = bundles[2 * r + 1];
  const float* ilu = acc_il + (size_t)u * DD;
  const float* blu = acc_bl + (size_t)u * DD;
  const float* i0  = ilb + (size_t)b0 * DD;
  const float* i1  = ilb + (size_t)b1 * DD;
  const float* c0  = acc_bl + (size_t)(NU + b0) * DD;
  const float* c1  = acc_bl + (size_t)(NU + b1) * DD;
  float p0 = 0.f, p1 = 0.f;
  #pragma unroll 8
  for (int j = 0; j < DD; ++j) {
    p0 += ilu[j] * i0[j] + blu[j] * c0[j];
    p1 += ilu[j] * i1[j] + blu[j] * c1[j];
  }
  float x = p0 - p1;
  // -log_sigmoid(x), numerically stable
  float nls = fmaxf(-x, 0.0f) + log1pf(expf(-fabsf(x)));
  atomic_add_f32(s_bpr, nls);
}

// ---------------- gather + L2 normalize + f16 pack + positive score ----------------
__global__ void k_gather_norm(const int* __restrict__ idx, int idxStride,
                              const float* __restrict__ baseP, const float* __restrict__ baseA,
                              float* __restrict__ Pn, float* __restrict__ An,
                              _Float16* __restrict__ Ph, _Float16* __restrict__ Ah,
                              float* __restrict__ pos, int B) {
  int r = blockIdx.x * blockDim.x + threadIdx.x;
  if (r >= B) return;
  int ip = idx[(size_t)r * idxStride];
  const float* prow = baseP + (size_t)ip * DD;
  const float* arow = baseA + (size_t)ip * DD;
  float np = 0.f, na = 0.f;
  for (int j = 0; j < DD; ++j) { np += prow[j] * prow[j]; na += arow[j] * arow[j]; }
  float ivp = 1.0f / fmaxf(sqrtf(np), 1e-12f);
  float iva = 1.0f / fmaxf(sqrtf(na), 1e-12f);
  float dot = 0.f;
  for (int j = 0; j < DD; ++j) {
    float pv = prow[j] * ivp;
    float av = arow[j] * iva;
    Pn[(size_t)r * DD + j] = pv;
    An[(size_t)r * DD + j] = av;
    Ph[(size_t)r * DD + j] = (_Float16)pv;
    Ah[(size_t)r * DD + j] = (_Float16)av;
    dot += pv * av;
  }
  pos[r] = dot * INV_CTEMP;
}

// ---------------- S = (P @ A^T) / C_TEMP  via v_wmma_f32_16x16x32_f16 ----------------
// 2048x2048x64. One wave per 16x32 output strip (two N-tiles) so the A fragment is
// reused across two WMMAs per K step. 8 waves/block, exact grid, EXEC all-ones.
__global__ __launch_bounds__(256) void k_gemm_wmma(const _Float16* __restrict__ P,
                                                   const _Float16* __restrict__ A,
                                                   float* __restrict__ S, float scale) {
  const int N = BB;
  const int TCP = N / 32;                   // 64 column-pair strips per row of tiles
  int wave = threadIdx.x >> 5;
  int lane = threadIdx.x & 31;
  int t = blockIdx.x * 8 + wave;            // 128 * 64 = 8192 wave-strips
  int tr = t / TCP, tc = t % TCP;
  int row0 = tr * 16, col0 = tc * 32;
  int mn = lane & 15;                       // M (for A-frag) / N (for B,D)
  int h  = lane >> 4;                       // lane half-group

  const _Float16* prow  = P + (size_t)(row0 + mn) * DD;
  const _Float16* arow0 = A + (size_t)(col0 + mn) * DD;
  const _Float16* arow1 = A + (size_t)(col0 + 16 + mn) * DD;

  v8f c0 = {}, c1 = {};
  #pragma unroll
  for (int k0 = 0; k0 < DD; k0 += 32) {
    v16h af, bf0, bf1;
    // A 16x32 f16 layout: lanes 0-15 -> K 0-15; lanes 16-31 -> K 16-31 (matches B)
    #pragma unroll
    for (int j = 0; j < 16; ++j) {
      af[j] = prow[k0 + h * 16 + j];
    }
    // B 32x16 f16 layout: lanes 0-15 -> K 0-15; lanes 16-31 -> K 16-31; N = lane%16
    #pragma unroll
    for (int j = 0; j < 16; ++j) {
      bf0[j] = arow0[k0 + h * 16 + j];
      bf1[j] = arow1[k0 + h * 16 + j];
    }
    c0 = __builtin_amdgcn_wmma_f32_16x16x32_f16(false, af, false, bf0,
                                                (short)0, c0, false, false);
    c1 = __builtin_amdgcn_wmma_f32_16x16x32_f16(false, af, false, bf1,
                                                (short)0, c1, false, false);
  }
  // D f32 layout: lanes 0-15 -> M = r; lanes 16-31 -> M = 8 + r; N = lane%16
  float* srow = S + (size_t)(row0 + h * 8) * N + col0 + mn;
  #pragma unroll
  for (int r = 0; r < 8; ++r) {
    srow[(size_t)r * N]      = c0[r] * scale;
    srow[(size_t)r * N + 16] = c1[r] * scale;
  }
}

// ---------------- row-wise logsumexp + contrastive partial sum ----------------
__global__ void k_lse(const float* __restrict__ S, const float* __restrict__ pos,
                      float* __restrict__ s_c) {
  int row  = (blockIdx.x * blockDim.x + threadIdx.x) >> 5;
  int lane = threadIdx.x & 31;
  if (row >= BB) return;
  const float* r = S + (size_t)row * BB;
  float mx = -3.4e38f;
  for (int c = lane; c < BB; c += 32) mx = fmaxf(mx, r[c]);
  #pragma unroll
  for (int m = 16; m > 0; m >>= 1) mx = fmaxf(mx, __shfl_xor(mx, m, 32));
  float sum = 0.f;
  for (int c = lane; c < BB; c += 32) sum += expf(r[c] - mx);
  #pragma unroll
  for (int m = 16; m > 0; m >>= 1) sum += __shfl_xor(sum, m, 32);
  if (lane == 0) atomic_add_f32(s_c, (mx + logf(sum)) - pos[row]);
}

// ---------------- final combine ----------------
__global__ void k_final(const float* __restrict__ scal, float* __restrict__ out) {
  out[0] = scal[0] * (1.0f / BB);                       // bpr loss
  out[1] = 0.5f * (scal[1] + scal[2]) * (1.0f / BB);    // contrastive loss
}

// ---------------- workspace layout (float offsets) ----------------
#define F_ACC_IL  ((size_t)0)
#define F_F       (F_ACC_IL + (size_t)NIL * DD)
#define F_G       (F_F      + (size_t)NIL * DD)
#define F_ACC_BL  (F_G      + (size_t)NIL * DD)
#define F_ILB     (F_ACC_BL + (size_t)NBL * DD)
#define F_DINV_IL (F_ILB    + (size_t)NBU * DD)
#define F_DINV_BL (F_DINV_IL + (size_t)NIL)
#define F_WINV    (F_DINV_BL + (size_t)NBL)
#define F_S       (F_WINV   + (size_t)NBU)
#define F_PN_U    (F_S      + (size_t)BB * BB)
#define F_AN_U    (F_PN_U   + (size_t)BB * DD)
#define F_PN_B    (F_AN_U   + (size_t)BB * DD)
#define F_AN_B    (F_PN_B   + (size_t)BB * DD)
#define F_H       (F_AN_B   + (size_t)BB * DD)   // 4*BB*DD halves = 2*BB*DD floats
#define F_POS_U   (F_H      + (size_t)2 * BB * DD)
#define F_POS_B   (F_POS_U  + (size_t)BB)
#define F_SCAL    (F_POS_B  + (size_t)BB)

extern "C" void kernel_launch(void* const* d_in, const int* in_sizes, int n_in,
                              void* d_out, int out_size, void* d_ws, size_t ws_size,
                              hipStream_t stream) {
  const int*   users    = (const int*)d_in[0];
  const int*   bundles  = (const int*)d_in[1];
  const int*   ui_u     = (const int*)d_in[2];
  const int*   ui_i     = (const int*)d_in[3];
  const int*   ub_u     = (const int*)d_in[4];
  const int*   ub_b     = (const int*)d_in[5];
  const int*   bi_b     = (const int*)d_in[6];
  const int*   bi_i     = (const int*)d_in[7];
  const float* uf       = (const float*)d_in[8];
  const float* bf       = (const float*)d_in[9];
  const float* itf      = (const float*)d_in[10];
  const int E_ui = in_sizes[2], E_ub = in_sizes[4], E_bi = in_sizes[6];
  const int B    = in_sizes[0];

  float* ws = (float*)d_ws;
  float* acc_il  = ws + F_ACC_IL;
  float* f       = ws + F_F;
  float* g       = ws + F_G;
  float* acc_bl  = ws + F_ACC_BL;
  float* ilb     = ws + F_ILB;
  float* dinv_il = ws + F_DINV_IL;
  float* dinv_bl = ws + F_DINV_BL;
  float* winv    = ws + F_WINV;
  float* S       = ws + F_S;
  float* Pn_u    = ws + F_PN_U;
  float* An_u    = ws + F_AN_U;
  float* Pn_b    = ws + F_PN_B;
  float* An_b    = ws + F_AN_B;
  _Float16* Hb   = (_Float16*)(ws + F_H);
  _Float16* Ph_u = Hb;
  _Float16* Ah_u = Hb + (size_t)BB * DD;
  _Float16* Ph_b = Hb + (size_t)2 * BB * DD;
  _Float16* Ah_b = Hb + (size_t)3 * BB * DD;
  float* pos_u   = ws + F_POS_U;
  float* pos_b   = ws + F_POS_B;
  float* scal    = ws + F_SCAL;

  const int T = 256;
  // ---------- IL graph (user-item, n = NIL) ----------
  hipMemsetAsync(dinv_il, 0, (size_t)NIL * 4, stream);
  k_deg_pair<<<(E_ui + T - 1) / T, T, 0, stream>>>(ui_u, ui_i, E_ui, NU, dinv_il);
  k_dinv<<<(NIL + T - 1) / T, T, 0, stream>>>(dinv_il, NIL);
  hipMemcpyAsync(f, uf, (size_t)NU * DD * 4, hipMemcpyDeviceToDevice, stream);
  hipMemcpyAsync(f + (size_t)NU * DD, itf, (size_t)NI * DD * 4, hipMemcpyDeviceToDevice, stream);
  hipMemcpyAsync(acc_il, uf, (size_t)NU * DD * 4, hipMemcpyDeviceToDevice, stream);
  hipMemcpyAsync(acc_il + (size_t)NU * DD, itf, (size_t)NI * DD * 4, hipMemcpyDeviceToDevice, stream);
  for (int i = 0; i < 2; ++i) {
    hipMemsetAsync(g, 0, (size_t)NIL * DD * 4, stream);
    long long tot = (long long)E_ui * 32;
    k_spmm<<<(unsigned)((tot + T - 1) / T), T, 0, stream>>>(ui_u, ui_i, E_ui, NU, dinv_il, f, g);
    k_row_update<<<(NIL + 7) / 8, T, 0, stream>>>(g, f, acc_il, NIL, 1.0f / (float)(i + 2));
  }
  // ---------- bundle aggregation ----------
  hipMemsetAsync(winv, 0, (size_t)NBU * 4, stream);
  k_deg_single<<<(E_bi + T - 1) / T, T, 0, stream>>>(bi_b, E_bi, winv);
  k_winv<<<(NBU + T - 1) / T, T, 0, stream>>>(winv, NBU);
  hipMemsetAsync(ilb, 0, (size_t)NBU * DD * 4, stream);
  {
    long long tot = (long long)E_bi * 32;
    k_bundle_agg<<<(unsigned)((tot + T - 1) / T), T, 0, stream>>>(
        bi_b, bi_i, E_bi, winv, acc_il + (size_t)NU * DD, ilb);
  }
  // ---------- BL graph (user-bundle, n = NBL); reuses f/g ----------
  hipMemsetAsync(dinv_bl, 0, (size_t)NBL * 4, stream);
  k_deg_pair<<<(E_ub + T - 1) / T, T, 0, stream>>>(ub_u, ub_b, E_ub, NU, dinv_bl);
  k_dinv<<<(NBL + T - 1) / T, T, 0, stream>>>(dinv_bl, NBL);
  hipMemcpyAsync(f, uf, (size_t)NU * DD * 4, hipMemcpyDeviceToDevice, stream);
  hipMemcpyAsync(f + (size_t)NU * DD, bf, (size_t)NBU * DD * 4, hipMemcpyDeviceToDevice, stream);
  hipMemcpyAsync(acc_bl, uf, (size_t)NU * DD * 4, hipMemcpyDeviceToDevice, stream);
  hipMemcpyAsync(acc_bl + (size_t)NU * DD, bf, (size_t)NBU * DD * 4, hipMemcpyDeviceToDevice, stream);
  for (int i = 0; i < 2; ++i) {
    hipMemsetAsync(g, 0, (size_t)NBL * DD * 4, stream);
    long long tot = (long long)E_ub * 32;
    k_spmm<<<(unsigned)((tot + T - 1) / T), T, 0, stream>>>(ub_u, ub_b, E_ub, NU, dinv_bl, f, g);
    k_row_update<<<(NBL + 7) / 8, T, 0, stream>>>(g, f, acc_bl, NBL, 1.0f / (float)(i + 2));
  }
  // ---------- losses ----------
  hipMemsetAsync(scal, 0, 3 * 4, stream);
  k_bpr<<<(B + T - 1) / T, T, 0, stream>>>(users, bundles, acc_il, ilb, acc_bl, B, &scal[0]);
  k_gather_norm<<<(B + T - 1) / T, T, 0, stream>>>(users, 1, acc_il, acc_bl,
                                                   Pn_u, An_u, Ph_u, Ah_u, pos_u, B);
  k_gather_norm<<<(B + T - 1) / T, T, 0, stream>>>(bundles, 2, ilb, acc_bl + (size_t)NU * DD,
                                                   Pn_b, An_b, Ph_b, Ah_b, pos_b, B);
  // users contrastive: S = Pn_u @ An_u^T / C_TEMP, then logsumexp
  k_gemm_wmma<<<(BB / 16) * (BB / 32) / 8, T, 0, stream>>>(Ph_u, Ah_u, S, INV_CTEMP);
  k_lse<<<BB / 8, T, 0, stream>>>(S, pos_u, &scal[1]);
  // bundles contrastive (reuse S)
  k_gemm_wmma<<<(BB / 16) * (BB / 32) / 8, T, 0, stream>>>(Ph_b, Ah_b, S, INV_CTEMP);
  k_lse<<<BB / 8, T, 0, stream>>>(S, pos_b, &scal[2]);
  k_final<<<1, 1, 0, stream>>>(scal, (float*)d_out);
}
